// STFTModule_47906065219847
// MI455X (gfx1250) — matrix operand order repaired and checked by reading, not compiled, not run on hardware
//
#include <hip/hip_runtime.h>

#define NS    262144      // samples per batch
#define NFFT  1024
#define HOP   256
#define FB    513         // frequency bins
#define NFRM  1025        // frames
#define NBAT  16

typedef __attribute__((ext_vector_type(16))) _Float16 v16h;
typedef __attribute__((ext_vector_type(8)))  _Float16 v8h;
typedef __attribute__((ext_vector_type(8)))  float    v8f;

#define WMMA(A,B,C) __builtin_amdgcn_wmma_f32_16x16x32_f16(false,(A),false,(B),(short)0,(C),false,false)

__device__ __forceinline__ v16h frag16(const _Float16* p0, const _Float16* p1) {
  v8h a = *(const v8h*)p0;
  v8h b = *(const v8h*)p1;
  return __builtin_shufflevector(a, b, 0,1,2,3,4,5,6,7,8,9,10,11,12,13,14,15);
}

__device__ __forceinline__ int reflect_idx(int j) {
  j = (j < 0) ? -j : j;
  j = (j >= NS) ? (2*NS - 2 - j) : j;
  return j;
}

// One stage buffer, halfword units relative to buffer base:
//   A_hi [0,5120)  A_lo [5120,10240)  B_hi [10240,15360)  B_lo [15360,20480)
// A and B tiles are [128][40] halves (pad keeps 16B alignment, conflict-free banks).
#define BUF_HALFS 20480

// Block tile: 128 frames (M) x 128 channels (N), K chunk = 32, double-buffered.
// Channels 0..63 -> real part of bins f0..f0+63 ; 64..127 -> imag part.
// 8 waves as 2(M) x 4(N); each wave: 64x32 via 4x2 WMMA tiles (8 accumulators).
template <bool SAFE>
__global__ __launch_bounds__(256) void stft_wmma(const float* __restrict__ sig_all,
                                                 const float* __restrict__ kern,
                                                 float* __restrict__ out) {
  __shared__ __align__(16) _Float16 smem[2 * BUF_HALFS];   // 81920 B
  float* sOut = (float*)smem;                              // reused: [64][128] float2 = 65536 B

  const int tid  = threadIdx.x;
  const int lane = tid & 31;
  const int widx = tid >> 5;
  const int wm   = widx >> 2;   // 0..1
  const int wn   = widx & 3;    // 0..3
  const int lh   = lane & 15;
  const int lp   = lane >> 4;

  const int fblk = blockIdx.x;                                   // 0..8
  const int mblk = SAFE ? (blockIdx.y + 1)                       // 1..6 (no reflect possible)
                        : (blockIdx.y == 0 ? 0 : blockIdx.y + 6);// {0,7,8}
  const int b    = blockIdx.z;
  const int tBase = mblk * 128;
  const int f0    = fblk * 64;

  const float* sig = sig_all + (size_t)b * NS;

  // staging coords: 16 rows per pass, 16 threads x 2 contiguous k per row
  const int srow16 = tid >> 4;        // 0..15
  const int skk    = (tid & 15) * 2;  // 0..30

  int aOff[8];
#pragma unroll
  for (int p = 0; p < 8; ++p)
    aOff[p] = (tBase + p*16 + srow16) * HOP + skk - (NFFT/2);

  const float* bPtr[8];
#pragma unroll
  for (int p = 0; p < 8; ++p) {
    int row = p*16 + srow16;            // 0..127
    int fl = row & 63, prt = row >> 6;
    int c = prt ? (FB + f0 + fl) : (f0 + fl);
    c = (c > 1025) ? 1025 : c;          // clamp OOB imag rows; stores are guarded
    bPtr[p] = kern + (size_t)c * NFFT + skk;
  }

  // fragment LDS offsets (halfword units, relative to buffer base)
  const int cB0 = wn*32 + lh;
  const int cB1 = cB0 + 16;
  const int oB0 = cB0*40 + lp*8;        // + {0,16}
  const int oB1 = cB1*40 + lp*8;
  int oA[4];
#pragma unroll
  for (int i = 0; i < 4; ++i)
    oA[i] = (wm*64 + i*16 + lh)*40 + lp*8;   // + {0,16}

  v8f acc[4][2];
  {
    v8f zero = {};
#pragma unroll
    for (int i = 0; i < 4; ++i) { acc[i][0] = zero; acc[i][1] = zero; }
  }

  float ra[8][2], rb[8][2];             // raw f32 pipeline registers

  auto loadChunk = [&](int k0) {
    if (SAFE) {
#pragma unroll
      for (int p = 0; p < 8; ++p) {
        float2 v = *(const float2*)(sig + (aOff[p] + k0));
        ra[p][0] = v.x; ra[p][1] = v.y;
      }
    } else {
#pragma unroll
      for (int p = 0; p < 8; ++p) {
        int jb = aOff[p] + k0;
        ra[p][0] = sig[reflect_idx(jb)];
        ra[p][1] = sig[reflect_idx(jb + 1)];
      }
    }
#pragma unroll
    for (int p = 0; p < 8; ++p) {
      float2 v = *(const float2*)(bPtr[p] + k0);
      rb[p][0] = v.x; rb[p][1] = v.y;
    }
  };

  auto cvtStore = [&](_Float16* base) {
    _Float16* Ahi = base;
    _Float16* Alo = base + 5120;
    _Float16* Bhi = base + 10240;
    _Float16* Blo = base + 15360;
#pragma unroll
    for (int p = 0; p < 8; ++p) {
      int o = (p*16 + srow16)*40 + skk;
      _Float16 h0 = (_Float16)ra[p][0];
      _Float16 h1 = (_Float16)ra[p][1];
      Ahi[o]   = h0;                       Ahi[o+1] = h1;
      Alo[o]   = (_Float16)(ra[p][0] - (float)h0);
      Alo[o+1] = (_Float16)(ra[p][1] - (float)h1);
    }
#pragma unroll
    for (int p = 0; p < 8; ++p) {
      int o = (p*16 + srow16)*40 + skk;
      _Float16 h0 = (_Float16)rb[p][0];
      _Float16 h1 = (_Float16)rb[p][1];
      Bhi[o]   = h0;                       Bhi[o+1] = h1;
      Blo[o]   = (_Float16)(rb[p][0] - (float)h0);
      Blo[o+1] = (_Float16)(rb[p][1] - (float)h1);
    }
  };

  auto compute = [&](const _Float16* base) {
    const _Float16* Ahi = base;
    const _Float16* Alo = base + 5120;
    const _Float16* Bhi = base + 10240;
    const _Float16* Blo = base + 15360;
    // B fragments reused across all four M-subtiles
    v16h b0h = frag16(Bhi + oB0, Bhi + oB0 + 16);
    v16h b0l = frag16(Blo + oB0, Blo + oB0 + 16);
    v16h b1h = frag16(Bhi + oB1, Bhi + oB1 + 16);
    v16h b1l = frag16(Blo + oB1, Blo + oB1 + 16);
#pragma unroll
    for (int i = 0; i < 4; ++i) {
      v16h ah = frag16(Ahi + oA[i], Ahi + oA[i] + 16);
      v16h al = frag16(Alo + oA[i], Alo + oA[i] + 16);
      // split-f16 fp32-accurate product: hi*hi + hi*lo + lo*hi
      acc[i][0] = WMMA(ah, b0h, acc[i][0]);
      acc[i][1] = WMMA(ah, b1h, acc[i][1]);
      acc[i][0] = WMMA(ah, b0l, acc[i][0]);
      acc[i][1] = WMMA(ah, b1l, acc[i][1]);
      acc[i][0] = WMMA(al, b0h, acc[i][0]);
      acc[i][1] = WMMA(al, b1h, acc[i][1]);
    }
  };

  // ---- software-pipelined, double-buffered K loop ----
  loadChunk(0);
  cvtStore(smem);                         // buf0 <- chunk 0
  for (int k0 = 0; k0 < NFFT; k0 += 64) {
    __syncthreads();                      // buf0 ready
    loadChunk(k0 + 32);                   // global loads overlap WMMAs below
    compute(smem);                        // chunk k0 from buf0
    cvtStore(smem + BUF_HALFS);           // buf1 <- chunk k0+32 (waits loads here)
    __syncthreads();                      // buf1 ready
    const bool more = (k0 + 64) < NFFT;
    if (more) loadChunk(k0 + 64);
    compute(smem + BUF_HALFS);            // chunk k0+32 from buf1
    if (more) cvtStore(smem);             // buf0 <- chunk k0+64
  }

  __syncthreads();

  // ---- stage D to LDS as [f_local(64)][t_local(128)] float2(re,im) ----
  {
    const int tB0 = wm*64 + lp*8;
#pragma unroll
    for (int j = 0; j < 2; ++j) {
      int colj = cB0 + j*16;              // 0..127
      int flc = colj & 63;
      int prt = colj >> 6;
#pragma unroll
      for (int i = 0; i < 4; ++i) {
        v8f a = acc[i][j];
#pragma unroll
        for (int r = 0; r < 8; ++r) {
          int tl = tB0 + i*16 + r;
          sOut[(flc*128 + tl)*2 + prt] = a[r];
        }
      }
    }
  }
  __syncthreads();

  // ---- contiguous float2 stores along t ----
  {
    float2* o2 = (float2*)out;
    const float2* s2 = (const float2*)sOut;
    const int ts = tid & 7;    // 0..7
#pragma unroll
    for (int fp = 0; fp < 2; ++fp) {
      int fi = fp*32 + (tid >> 3);        // 0..63
      int f  = f0 + fi;
      if (f < FB) {
        size_t rowbase = ((size_t)b*FB + f) * (size_t)NFRM;
#pragma unroll
        for (int s = 0; s < 16; ++s) {
          int tl = ts + s*8;
          int t = tBase + tl;
          if (t < NFRM) o2[rowbase + t] = s2[fi*128 + tl];
        }
      }
    }
  }
}

extern "C" void kernel_launch(void* const* d_in, const int* in_sizes, int n_in,
                              void* d_out, int out_size, void* d_ws, size_t ws_size,
                              hipStream_t stream) {
  (void)in_sizes; (void)n_in; (void)d_ws; (void)ws_size; (void)out_size;
  const float* sig  = (const float*)d_in[0];   // (16, 262144) f32
  const float* kern = (const float*)d_in[1];   // (1026, 1024) f32
  float* out = (float*)d_out;                  // (16, 513, 1025, 2) f32
  // hot kernel: m-blocks 1..6 (no reflection possible -> pure float2 loads)
  stft_wmma<true ><<<dim3(9, 6, NBAT), 256, 0, stream>>>(sig, kern, out);
  // edge kernel: m-blocks {0,7,8} (reflect-indexed loads)
  stft_wmma<false><<<dim3(9, 3, NBAT), 256, 0, stream>>>(sig, kern, out);
}
